// Attention_24189255811127
// MI455X (gfx1250) — matrix operand order repaired and checked by reading, not compiled, not run on hardware
//
#include <hip/hip_runtime.h>

#define HIDDEN 1024
#define SEQ    2048
#define BATCH  4
#define HEADS  16
#define HD     64
#define QKV3   (3 * HIDDEN)

typedef __attribute__((ext_vector_type(16))) __bf16 v16bf;
typedef __attribute__((ext_vector_type(8)))  __bf16 v8bf;
typedef __attribute__((ext_vector_type(4)))  __bf16 v4bf;
typedef __attribute__((ext_vector_type(8)))  float  v8f;

static __device__ __forceinline__ v16bf cat8(v8bf lo, v8bf hi) {
  return __builtin_shufflevector(lo, hi, 0,1,2,3,4,5,6,7,8,9,10,11,12,13,14,15);
}
static __device__ __forceinline__ v4bf cvt4(float4 f) {
  v4bf r; r[0] = (__bf16)f.x; r[1] = (__bf16)f.y; r[2] = (__bf16)f.z; r[3] = (__bf16)f.w;
  return r;
}
static __device__ __forceinline__ v8f wmma_bf16(v16bf a, v16bf b, v8f c) {
  return __builtin_amdgcn_wmma_f32_16x16x32_bf16(false, a, false, b, (short)0, c, false, false);
}

// ---------------------------------------------------------------------------
// Generic GEMM:  C[M,N] = A[M,K] @ W[K,N] + bias[N]   (fp32 in/out, bf16 WMMA)
// block = 128 threads (4 waves); each block computes a 64x64 tile; K-step 32.
// ---------------------------------------------------------------------------
#define AST 40   // LDS row stride (bf16 elems) for A tile [64][32]  (80B, 16B-mult)
#define BST 40   // LDS row stride for W tile stored transposed [64 n][32 k]

__global__ __launch_bounds__(128)
void gemm_bias_kernel(const float* __restrict__ A, const float* __restrict__ W,
                      const float* __restrict__ bias, float* __restrict__ C,
                      int M, int K, int N) {
  __shared__ __align__(16) __bf16 As[64 * AST];
  __shared__ __align__(16) __bf16 Bs[64 * BST];

  const int tid  = threadIdx.x;
  const int lane = tid & 31;
  const int wv   = tid >> 5;
  const int half = lane >> 4;   // 0: lanes 0-15, 1: lanes 16-31
  const int l16  = lane & 15;
  const int r0   = blockIdx.y * 64;
  const int c0   = blockIdx.x * 64;

  v8f acc[4];
  const v8f zf = {0.f,0.f,0.f,0.f,0.f,0.f,0.f,0.f};
  acc[0] = zf; acc[1] = zf; acc[2] = zf; acc[3] = zf;

  const int kbA = half ? 16 : 0;   // A-fragment K base (contiguous k = 16*half + s)
  const int kbB = half ? 16 : 0;   // B-fragment K base

  for (int k0 = 0; k0 < K; k0 += 32) {
    // Stage A tile: 64x32 fp32 -> bf16 LDS (row-major).  512 float4 chunks.
#pragma unroll
    for (int it = 0; it < 4; ++it) {
      int c   = tid + it * 128;
      int row = c >> 3;            // 8 float4 per 32-elem row
      int kc  = (c & 7) << 2;
      float4 f = *(const float4*)(A + (size_t)(r0 + row) * K + k0 + kc);
      *(v4bf*)(As + row * AST + kc) = cvt4(f);
    }
    // Stage W tile: 32x64 fp32 -> bf16 LDS, transposed to [n][k].
#pragma unroll
    for (int it = 0; it < 4; ++it) {
      int c  = tid + it * 128;
      int kr = c >> 4;             // 16 float4 per 64-elem row
      int nc = (c & 15) << 2;
      float4 f = *(const float4*)(W + (size_t)(k0 + kr) * N + c0 + nc);
      Bs[(nc + 0) * BST + kr] = (__bf16)f.x;
      Bs[(nc + 1) * BST + kr] = (__bf16)f.y;
      Bs[(nc + 2) * BST + kr] = (__bf16)f.z;
      Bs[(nc + 3) * BST + kr] = (__bf16)f.w;
    }
    __syncthreads();

    // A fragment: row = wave's 16-row strip + lane%16
    const __bf16* ap = As + (wv * 16 + l16) * AST;
    v16bf a = cat8(*(const v8bf*)(ap + kbA), *(const v8bf*)(ap + kbA + 8));

#pragma unroll
    for (int j = 0; j < 4; ++j) {
      const __bf16* bp = Bs + (j * 16 + l16) * BST + kbB;
      v16bf b = cat8(*(const v8bf*)bp, *(const v8bf*)(bp + 8));
      acc[j] = wmma_bf16(a, b, acc[j]);
    }
    __syncthreads();
  }

  // Epilogue: bias + store (C/D layout: vgpr i -> row i + 8*half, col = lane%16)
#pragma unroll
  for (int j = 0; j < 4; ++j) {
    int   col = c0 + j * 16 + l16;
    float bv  = bias[col];
#pragma unroll
    for (int i = 0; i < 8; ++i) {
      int row = r0 + wv * 16 + i + half * 8;
      C[(size_t)row * N + col] = acc[j][i] + bv;
    }
  }
}

// ---------------------------------------------------------------------------
// Flash attention: one workgroup = one (b,h), 64 query rows (16 per wave).
// Online softmax; K/V staged in LDS per 32-key tile; P re-laid out via LDS.
// qkv layout: qkv[b, n, s*1024 + h*64 + d], s in {0:q, 1:k, 2:v}
// out layout: out[b, n, h*64 + d]  (== [B*N, HIDDEN] for the proj GEMM)
// ---------------------------------------------------------------------------
#define KST 72   // Ks row stride [32 m][64 d]  (144B)
#define VST 40   // Vs row stride [64 d][32 m]  (80B)
#define PST 40   // P  row stride [16 r][32 m] per wave

__global__ __launch_bounds__(128)
void flash_attn_kernel(const float* __restrict__ qkv, float* __restrict__ out) {
  __shared__ __align__(16) __bf16 Ks[32 * KST];
  __shared__ __align__(16) __bf16 Vs[HD * VST];
  __shared__ __align__(16) __bf16 Ps[4 * 16 * PST];

  const int tid  = threadIdx.x;
  const int lane = tid & 31;
  const int wv   = tid >> 5;
  const int half = lane >> 4;
  const int l16  = lane & 15;
  const int q0   = blockIdx.x * 64;
  const int b    = blockIdx.y / HEADS;
  const int h    = blockIdx.y % HEADS;

  const float* base = qkv + (size_t)b * SEQ * QKV3;
  const int kbA = half ? 16 : 0;
  const int kbB = half ? 16 : 0;

  // Q fragments for this wave's 16 rows: two d-chunks of 32 (A layout).
  const int    qrow = q0 + wv * 16 + l16;
  const float* qp   = base + (size_t)qrow * QKV3 + h * HD;
  v16bf aq[2];
#pragma unroll
  for (int dc = 0; dc < 2; ++dc) {
    v16bf t;
#pragma unroll
    for (int e = 0; e < 8; ++e) t[e]     = (__bf16)qp[dc * 32 + kbA + e];
#pragma unroll
    for (int e = 0; e < 8; ++e) t[8 + e] = (__bf16)qp[dc * 32 + kbA + 8 + e];
    aq[dc] = t;
  }

  const v8f zf = {0.f,0.f,0.f,0.f,0.f,0.f,0.f,0.f};
  v8f   o[4] = {zf, zf, zf, zf};
  float m_i[8], l_i[8];
#pragma unroll
  for (int i = 0; i < 8; ++i) { m_i[i] = -1e30f; l_i[i] = 0.f; }

  const float scale = 0.125f;  // 1/sqrt(64)

  for (int j0 = 0; j0 < SEQ; j0 += 32) {
    // Cooperative K/V tile staging: 32 keys x 64 d, fp32 -> bf16.
#pragma unroll
    for (int it = 0; it < 4; ++it) {
      int c   = tid + it * 128;     // 512 float4 chunks
      int row = c >> 4;
      int dc4 = (c & 15) << 2;
      const float* kp = base + (size_t)(j0 + row) * QKV3 + HIDDEN + h * HD + dc4;
      *(v4bf*)(Ks + row * KST + dc4) = cvt4(*(const float4*)kp);
      const float* vp = base + (size_t)(j0 + row) * QKV3 + 2 * HIDDEN + h * HD + dc4;
      float4 g = *(const float4*)vp;
      Vs[(dc4 + 0) * VST + row] = (__bf16)g.x;   // transposed: [d][m]
      Vs[(dc4 + 1) * VST + row] = (__bf16)g.y;
      Vs[(dc4 + 2) * VST + row] = (__bf16)g.z;
      Vs[(dc4 + 3) * VST + row] = (__bf16)g.w;
    }
    __syncthreads();

    // S = Q @ K^T for two 16-key column tiles, contracting d in 2 chunks.
    v8f s0 = zf, s1 = zf;
#pragma unroll
    for (int dc = 0; dc < 2; ++dc) {
      const __bf16* p0 = Ks + l16 * KST + dc * 32 + kbB;
      s0 = wmma_bf16(aq[dc], cat8(*(const v8bf*)p0, *(const v8bf*)(p0 + 8)), s0);
      const __bf16* p1 = Ks + (16 + l16) * KST + dc * 32 + kbB;
      s1 = wmma_bf16(aq[dc], cat8(*(const v8bf*)p1, *(const v8bf*)(p1 + 8)), s1);
    }

    // Online softmax update (rows live across 16-lane half-waves).
#pragma unroll
    for (int i = 0; i < 8; ++i) {
      s0[i] *= scale; s1[i] *= scale;
      float v = fmaxf(s0[i], s1[i]);
      v = fmaxf(v, __shfl_xor(v, 1, 32));
      v = fmaxf(v, __shfl_xor(v, 2, 32));
      v = fmaxf(v, __shfl_xor(v, 4, 32));
      v = fmaxf(v, __shfl_xor(v, 8, 32));
      float mnew = fmaxf(m_i[i], v);
      float corr = __expf(m_i[i] - mnew);
      m_i[i] = mnew;
      float p0 = __expf(s0[i] - mnew);
      float p1 = __expf(s1[i] - mnew);
      s0[i] = p0; s1[i] = p1;
      float rs = p0 + p1;
      rs += __shfl_xor(rs, 1, 32);
      rs += __shfl_xor(rs, 2, 32);
      rs += __shfl_xor(rs, 4, 32);
      rs += __shfl_xor(rs, 8, 32);
      l_i[i] = l_i[i] * corr + rs;
#pragma unroll
      for (int t = 0; t < 4; ++t) o[t][i] *= corr;
    }

    // Re-layout P (C-layout) -> A-layout via this wave's LDS scratch.
    __bf16* pw = Ps + wv * 16 * PST;
#pragma unroll
    for (int i = 0; i < 8; ++i) {
      int row = i + half * 8;
      pw[row * PST + l16]      = (__bf16)s0[i];
      pw[row * PST + 16 + l16] = (__bf16)s1[i];
    }
    const __bf16* pr = pw + l16 * PST;
    v16bf pa = cat8(*(const v8bf*)(pr + kbA), *(const v8bf*)(pr + kbA + 8));

    // O += P @ V  (4 output d-tiles of 16)
#pragma unroll
    for (int t = 0; t < 4; ++t) {
      const __bf16* vb = Vs + (t * 16 + l16) * VST + kbB;
      o[t] = wmma_bf16(pa, cat8(*(const v8bf*)vb, *(const v8bf*)(vb + 8)), o[t]);
    }
    __syncthreads();
  }

  // Epilogue: normalize and store to [b, n, h*64 + d].
#pragma unroll
  for (int i = 0; i < 8; ++i) {
    float inv = 1.0f / l_i[i];
    int   row = q0 + wv * 16 + i + half * 8;
    float* op = out + ((size_t)b * SEQ + row) * HIDDEN + h * HD;
#pragma unroll
    for (int t = 0; t < 4; ++t) op[t * 16 + l16] = o[t][i] * inv;
  }
}

// ---------------------------------------------------------------------------
extern "C" void kernel_launch(void* const* d_in, const int* in_sizes, int n_in,
                              void* d_out, int out_size, void* d_ws, size_t ws_size,
                              hipStream_t stream) {
  (void)in_sizes; (void)n_in; (void)out_size; (void)ws_size;
  const float* x      = (const float*)d_in[0];
  const float* w_qkv  = (const float*)d_in[1];
  const float* b_qkv  = (const float*)d_in[2];
  const float* w_proj = (const float*)d_in[3];
  const float* b_proj = (const float*)d_in[4];
  float* out = (float*)d_out;

  float* qkv = (float*)d_ws;                          // [B*N, 3072]  (96 MB)
  float* att = qkv + (size_t)BATCH * SEQ * QKV3;      // [B*N, 1024]  (32 MB)

  const int M = BATCH * SEQ;
  dim3 blk(128);

  gemm_bias_kernel<<<dim3(QKV3 / 64, M / 64), blk, 0, stream>>>(
      x, w_qkv, b_qkv, qkv, M, HIDDEN, QKV3);

  flash_attn_kernel<<<dim3(SEQ / 64, BATCH * HEADS), blk, 0, stream>>>(qkv, att);

  gemm_bias_kernel<<<dim3(HIDDEN / 64, M / 64), blk, 0, stream>>>(
      att, w_proj, b_proj, out, M, HIDDEN, HIDDEN);
}
